// EMGEEGFusionEncoderv2_45217415692436
// MI455X (gfx1250) — compile-verified, matrix operand order checked by reading, not run on hardware
//
#include <hip/hip_runtime.h>
#include <hip/hip_bf16.h>

typedef __attribute__((ext_vector_type(16))) __bf16          v16bf;
typedef __attribute__((ext_vector_type(16))) unsigned short  v16u;
typedef __attribute__((ext_vector_type(8)))  float           v8f;
typedef __attribute__((ext_vector_type(4)))  unsigned int    v4u;
typedef __attribute__((ext_vector_type(8)))  int             v8i;
typedef __attribute__((ext_vector_type(4)))  int             v4i;

#define NNODE 2048
#define TDIM  1000
#define TPAD  1024
#define HDIM  512
#define LDIM  128
#define NEDGE 32768
#define GHDIM 256
#define NCAT  4096   // 2*NNODE

// ---------------- helpers ----------------
__device__ __forceinline__ unsigned short f2bf(float f) {
  unsigned int u = __float_as_uint(f);
  unsigned int r = u + 0x7FFFu + ((u >> 16) & 1u);   // round-to-nearest-even
  return (unsigned short)(r >> 16);
}

__device__ __forceinline__ void atomicMaxF(float* addr, float val) {
  int* ai = (int*)addr;
  int old = __float_as_int(*addr);
  while (val > __int_as_float(old)) {
    int assumed = old;
    old = atomicCAS(ai, assumed, __float_as_int(val));
    if (old == assumed) break;
  }
}

// f32[R,C] -> bf16[Rp,Cp], zero padding outside [R,C]
__global__ void k_cvt_pad(const float* __restrict__ src, unsigned short* __restrict__ dst,
                          int R, int C, int Rp, int Cp) {
  int i = blockIdx.x * blockDim.x + threadIdx.x;
  int total = Rp * Cp;
  if (i >= total) return;
  int r = i / Cp, c = i - r * Cp;
  dst[i] = (r < R && c < C) ? f2bf(src[r * C + c]) : (unsigned short)0;
}

__global__ void k_fill(float* __restrict__ p, int n, float v) {
  int i = blockIdx.x * blockDim.x + threadIdx.x;
  if (i < n) p[i] = v;
}

// ---------------- WMMA GEMM (TDM-staged A, LDS-transposed B) ----------------
// C[M,N] = A[M,K] @ B[K,N] (+bias)(+relu). A,B bf16 row-major, C f32.
// 256 threads = 8 waves; block tile 64(M) x 64(N); each wave owns 32x16 of C
// (2 accumulators sharing one B fragment -> 2 WMMA per barrier pair).
// Requires: M%64==0, N%64==0, K%32==0 (caller pads).
__global__ __launch_bounds__(256)
void k_gemm_bf16(const unsigned short* __restrict__ A, int lda,
                 const unsigned short* __restrict__ B, int ldb,
                 float* __restrict__ C, int ldc,
                 const float* __restrict__ bias, int relu, int K) {
  __shared__ unsigned short As[64 * 32];    // row-major [m][k]  (filled by TDM)
  __shared__ unsigned short Bst[64 * 32];   // n-major   [n][k]

  const int tid    = threadIdx.x;
  const int lane   = tid & 31;
  const int wave   = tid >> 5;
  const int waveMp = wave >> 2;             // 0..1 -> owns rows [waveMp*32, +32)
  const int waveN  = wave & 3;              // 0..3
  const int mBase  = blockIdx.y * 64;
  const int nBase  = blockIdx.x * 64;

  const int lrow = lane & 15;
  const int khi  = (lane & 16) >> 1;        // lanes 16..31 hold the K+8 halves

  v8f acc0 = {}, acc1 = {};

  for (int k0 = 0; k0 < K; k0 += 32) {
    // ---- stage A 64x32 tile via Tensor Data Mover (wave 0 issues; EXEC-ignoring op) ----
    if (tid < 32) {
      unsigned long long ga = (unsigned long long)(uintptr_t)(A + (size_t)mBase * lda + k0);
      unsigned ldsoff = (unsigned)(uintptr_t)As;   // flat LDS addr: low 32 bits = LDS byte offset
      // D# group0: count=1 | lds_addr | global_addr[56:0] | type=2
      v4u g0 = { 1u, ldsoff, (unsigned)ga,
                 (unsigned)((ga >> 32) & 0x01FFFFFFu) | (2u << 30) };
      // D# group1: data_size=2B; tensor_dim0=lda; tensor_dim1=big; tile 32x64; dim0_stride=lda
      unsigned td0 = (unsigned)lda;
      unsigned td1 = 1u << 20;
      v8i g1 = { (int)(1u << 16),
                 (int)((td0 & 0xFFFFu) << 16),
                 (int)(((td0 >> 16) & 0xFFFFu) | ((td1 & 0xFFFFu) << 16)),
                 (int)(((td1 >> 16) & 0xFFFFu) | (32u << 16)),
                 (int)64u,
                 (int)td0,
                 0, 0 };
      v4i gz = {};
#if __has_include(<hip/amd_detail/amd_gfx1250_TDM.h>)
      v8i gz8 = {};
      __builtin_amdgcn_tensor_load_to_lds(g0, g1, gz, gz, gz8, 0);
#else
      __builtin_amdgcn_tensor_load_to_lds(g0, g1, gz, gz, 0);
#endif
      __builtin_amdgcn_s_wait_tensorcnt((short)0);
    }
    // ---- stage B 32x64 tile transposed to n-major: 1024 dwords, 4 per thread ----
#pragma unroll
    for (int t = 0; t < 4; ++t) {
      int d  = tid + t * 256;
      int r  = d >> 5;
      int c2 = (d & 31) * 2;
      unsigned int v = *(const unsigned int*)&B[(k0 + r) * ldb + nBase + c2];
      Bst[(c2    ) * 32 + r] = (unsigned short)(v & 0xFFFFu);
      Bst[(c2 + 1) * 32 + r] = (unsigned short)(v >> 16);
    }
    if (k0 + 32 < K)   // prefetch next B tile (global_prefetch_b8)
      __builtin_prefetch(&B[(k0 + 32 + (tid & 31)) * ldb + nBase], 0, 1);
    __syncthreads();

    // ---- fragments per CDNA5 16-bit 16x32 layout (contiguous 8-elem runs -> ds_load_b128) ----
    v16u ub, ua0, ua1;
#pragma unroll
    for (int j = 0; j < 16; ++j) {
      int vg = j >> 1, p = j & 1;
      int k  = ((vg & 3) * 2 + p) + ((vg >> 2) * 16) + khi;
      ub[j]  = Bst[(waveN * 16 + lrow) * 32 + k];
      ua0[j] = As [(waveMp * 32 +      lrow) * 32 + k];
      ua1[j] = As [(waveMp * 32 + 16 + lrow) * 32 + k];
    }
    acc0 = __builtin_amdgcn_wmma_f32_16x16x32_bf16(
        false, __builtin_bit_cast(v16bf, ua0),
        false, __builtin_bit_cast(v16bf, ub),
        (short)0, acc0, false, false);
    acc1 = __builtin_amdgcn_wmma_f32_16x16x32_bf16(
        false, __builtin_bit_cast(v16bf, ua1),
        false, __builtin_bit_cast(v16bf, ub),
        (short)0, acc1, false, false);
    __syncthreads();
  }

  // ---- epilogue: D layout vgpr v -> M = v + 8*(lane>=16); N = lane&15 ----
#pragma unroll
  for (int t = 0; t < 2; ++t) {
    v8f acc = t ? acc1 : acc0;
#pragma unroll
    for (int vg = 0; vg < 8; ++vg) {
      int m   = vg + ((lane >> 4) << 3);
      int n   = lane & 15;
      int row = mBase + waveMp * 32 + t * 16 + m;
      int col = nBase + waveN * 16 + n;
      float v = acc[vg];
      if (bias) v += bias[col];
      if (relu) v = v > 0.f ? v : 0.f;
      C[row * ldc + col] = v;
    }
  }
}

// ---------------- GAT attention pieces ----------------
__global__ void k_gemv_att(const float* __restrict__ z, const float* __restrict__ avs,
                           const float* __restrict__ avd, float* __restrict__ as_,
                           float* __restrict__ ad_, int rows, int cols) {
  int row  = blockIdx.x * 8 + (threadIdx.x >> 5);
  int lane = threadIdx.x & 31;
  if (row >= rows) return;
  const float* zr = z + row * cols;
  float ss = 0.f, sd = 0.f;
  for (int c = lane; c < cols; c += 32) {
    float v = zr[c];
    ss += v * avs[c];
    sd += v * avd[c];
  }
  for (int o = 16; o > 0; o >>= 1) {
    ss += __shfl_down(ss, o, 32);
    sd += __shfl_down(sd, o, 32);
  }
  if (lane == 0) { as_[row] = ss; ad_[row] = sd; }
}

__global__ void k_edge_max(const int* __restrict__ src, const int* __restrict__ dst,
                           const float* __restrict__ as_, const float* __restrict__ ad_,
                           float* __restrict__ logit, float* __restrict__ mx,
                           int nE, int nN) {
  int i = blockIdx.x * blockDim.x + threadIdx.x;
  if (i >= nE + nN) return;
  int s, d;
  if (i < nE) { s = src[i]; d = dst[i]; } else { s = i - nE; d = s; }
  float e = as_[s] + ad_[d];
  e = e > 0.f ? e : 0.2f * e;            // leaky_relu(0.2)
  logit[i] = e;
  atomicMaxF(&mx[d], e);
}

__global__ void k_edge_exp(const int* __restrict__ dst, float* __restrict__ logit,
                           const float* __restrict__ mx, float* __restrict__ den,
                           int nE, int nN) {
  int i = blockIdx.x * blockDim.x + threadIdx.x;
  if (i >= nE + nN) return;
  int d = (i < nE) ? dst[i] : (i - nE);
  float e = expf(logit[i] - mx[d]);
  logit[i] = e;
  atomicAdd(&den[d], e);
}

// one block per edge; scatter alpha * z[s] into out[d]
__global__ void k_edge_agg(const int* __restrict__ src, const int* __restrict__ dst,
                           const float* __restrict__ ecoef, const float* __restrict__ den,
                           const float* __restrict__ z, float* __restrict__ out,
                           int nE, int F) {
  int i = blockIdx.x;
  int s, d;
  if (i < nE) { s = src[i]; d = dst[i]; } else { s = i - nE; d = s; }
  float alpha = ecoef[i] / (den[d] + 1e-16f);
  const float* zs = z + s * F;
  float*       od = out + d * F;
  for (int f = threadIdx.x; f < F; f += blockDim.x)
    atomicAdd(&od[f], alpha * zs[f]);
}

// out = agg + bias (optional relu); emit bf16 and/or f32
__global__ void k_finalize(const float* __restrict__ agg, const float* __restrict__ bias,
                           int relu, unsigned short* __restrict__ obf,
                           float* __restrict__ of32, int n, int C) {
  int i = blockIdx.x * blockDim.x + threadIdx.x;
  if (i >= n) return;
  float v = agg[i] + bias[i % C];
  if (relu) v = v > 0.f ? v : 0.f;
  if (obf)  obf[i]  = f2bf(v);
  if (of32) of32[i] = v;
}

// ---------------- GIN (mask is provably all-ones -> column sum) ----------------
__global__ void k_colsum(const float* __restrict__ x, float* __restrict__ S, int R, int C) {
  __shared__ float red[256];
  int col = blockIdx.x;
  float s = 0.f;
  for (int r = threadIdx.x; r < R; r += 256) s += x[r * C + col];
  red[threadIdx.x] = s;
  __syncthreads();
  for (int o = 128; o > 0; o >>= 1) {
    if (threadIdx.x < o) red[threadIdx.x] += red[threadIdx.x + o];
    __syncthreads();
  }
  if (threadIdx.x == 0) S[col] = red[0];
}

__global__ void k_addbcast_bf(const float* __restrict__ x, const float* __restrict__ S,
                              unsigned short* __restrict__ y, int n, int C) {
  int i = blockIdx.x * blockDim.x + threadIdx.x;
  if (i >= n) return;
  y[i] = f2bf(x[i] + S[i % C]);
}

// ---------------- host orchestration ----------------
extern "C" void kernel_launch(void* const* d_in, const int* in_sizes, int n_in,
                              void* d_out, int out_size, void* d_ws, size_t ws_size,
                              hipStream_t stream) {
  (void)in_sizes; (void)n_in; (void)out_size; (void)ws_size;

  // input order: emg_x, eeg_x, params(depth-first insertion order), emg_ei, eeg_ei
  const float* xin[2] = { (const float*)d_in[0], (const float*)d_in[1] };
  const int*   ein[2] = { (const int*)d_in[30], (const int*)d_in[31] };
  struct GatP { const float *W1,*as1,*ad1,*b1,*W2,*as2,*ad2,*b2,*Wp,*bp; };
  GatP gp[2];
  for (int m = 0; m < 2; ++m) {
    int b = 2 + m * 8;
    gp[m].W1 = (const float*)d_in[b+0]; gp[m].as1 = (const float*)d_in[b+1];
    gp[m].ad1 = (const float*)d_in[b+2]; gp[m].b1 = (const float*)d_in[b+3];
    gp[m].W2 = (const float*)d_in[b+4]; gp[m].as2 = (const float*)d_in[b+5];
    gp[m].ad2 = (const float*)d_in[b+6]; gp[m].b2 = (const float*)d_in[b+7];
    gp[m].Wp = (const float*)d_in[18 + m*2]; gp[m].bp = (const float*)d_in[19 + m*2];
  }
  const float* ginW1a = (const float*)d_in[22]; const float* ginB1a = (const float*)d_in[23];
  const float* ginW2a = (const float*)d_in[24]; const float* ginB2a = (const float*)d_in[25];
  const float* ginW1b = (const float*)d_in[26]; const float* ginB1b = (const float*)d_in[27];
  const float* ginW2b = (const float*)d_in[28]; const float* ginB2b = (const float*)d_in[29];
  float* out = (float*)d_out;

  // workspace carve-out
  char* wp = (char*)d_ws; size_t off = 0;
  auto take = [&](size_t bytes) -> void* {
    void* p = wp + off; off += (bytes + 255) & ~(size_t)255; return p;
  };
  unsigned short* bfA  = (unsigned short*)take((size_t)NNODE * TPAD * 2);   // 4 MB
  unsigned short* bfB  = (unsigned short*)take((size_t)TPAD  * HDIM * 2);   // 1 MB
  float* bufZ  = (float*)take((size_t)NCAT * GHDIM * 4);                    // 4 MB
  float* bufH  = (float*)take((size_t)NCAT * GHDIM * 4);                    // 4 MB
  float* bufC  = (float*)take((size_t)NCAT * LDIM  * 4);                    // 2 MB
  float* vas   = (float*)take(NNODE * 4);
  float* vad   = (float*)take(NNODE * 4);
  float* vmx   = (float*)take(NNODE * 4);
  float* vden  = (float*)take(NNODE * 4);
  float* ecoef = (float*)take((size_t)(NEDGE + NNODE) * 4);
  float* Svec  = (float*)take(GHDIM * 4);

  const int EB = (NEDGE + NNODE + 255) / 256;   // edge-parallel grid
  auto g1 = [](int n) { return dim3((unsigned)((n + 255) / 256)); };

  auto gemm = [&](const unsigned short* A, int lda, const unsigned short* B, int ldb,
                  float* C, int ldc, const float* bias, int relu, int M, int N, int K) {
    dim3 grid((unsigned)(N / 64), (unsigned)(M / 64));
    k_gemm_bf16<<<grid, 256, 0, stream>>>(A, lda, B, ldb, C, ldc, bias, relu, K);
  };
  auto gat_softmax_agg = [&](const int* ei, const float* z, float* agg, int F) {
    const int* src = ei; const int* dst = ei + NEDGE;
    k_fill<<<g1(NNODE), 256, 0, stream>>>(vmx, NNODE, -1e30f);
    k_fill<<<g1(NNODE), 256, 0, stream>>>(vden, NNODE, 0.f);
    k_fill<<<g1(NNODE * F), 256, 0, stream>>>(agg, NNODE * F, 0.f);
    k_edge_max<<<EB, 256, 0, stream>>>(src, dst, vas, vad, ecoef, vmx, NEDGE, NNODE);
    k_edge_exp<<<EB, 256, 0, stream>>>(dst, ecoef, vmx, vden, NEDGE, NNODE);
    k_edge_agg<<<dim3(NEDGE + NNODE), (F >= 256 ? 256 : F), 0, stream>>>(
        src, dst, ecoef, vden, z, agg, NEDGE, F);
  };

  // ---- per-modality GAT + projection (sequential, scratch reused) ----
  for (int m = 0; m < 2; ++m) {
    // layer 1: z1 = x @ W1   (K padded 1000 -> 1024)
    k_cvt_pad<<<g1(NNODE * TPAD), 256, 0, stream>>>(xin[m], bfA, NNODE, TDIM, NNODE, TPAD);
    k_cvt_pad<<<g1(TPAD * HDIM), 256, 0, stream>>>(gp[m].W1, bfB, TDIM, HDIM, TPAD, HDIM);
    gemm(bfA, TPAD, bfB, HDIM, bufZ, HDIM, nullptr, 0, NNODE, HDIM, TPAD);
    k_gemv_att<<<dim3(NNODE / 8), 256, 0, stream>>>(bufZ, gp[m].as1, gp[m].ad1, vas, vad, NNODE, HDIM);
    gat_softmax_agg(ein[m], bufZ, bufH, HDIM);
    // h1 = relu(agg + b1) -> bf16 (A of next GEMM)
    k_finalize<<<g1(NNODE * HDIM), 256, 0, stream>>>(bufH, gp[m].b1, 1, bfA, nullptr, NNODE * HDIM, HDIM);

    // layer 2: z2 = h1 @ W2
    k_cvt_pad<<<g1(HDIM * LDIM), 256, 0, stream>>>(gp[m].W2, bfB, HDIM, LDIM, HDIM, LDIM);
    gemm(bfA, HDIM, bfB, LDIM, bufZ, LDIM, nullptr, 0, NNODE, LDIM, HDIM);
    k_gemv_att<<<dim3(NNODE / 8), 256, 0, stream>>>(bufZ, gp[m].as2, gp[m].ad2, vas, vad, NNODE, LDIM);
    gat_softmax_agg(ein[m], bufZ, bufH, LDIM);
    // gat_out = agg + b2 -> bf16
    k_finalize<<<g1(NNODE * LDIM), 256, 0, stream>>>(bufH, gp[m].b2, 0, bfA, nullptr, NNODE * LDIM, LDIM);

    // projection into concat buffer rows [m*N .. m*N+N)
    k_cvt_pad<<<g1(LDIM * LDIM), 256, 0, stream>>>(gp[m].Wp, bfB, LDIM, LDIM, LDIM, LDIM);
    gemm(bfA, LDIM, bfB, LDIM, bufC + (size_t)m * NNODE * LDIM, LDIM, gp[m].bp, 0, NNODE, LDIM, LDIM);
  }

  // ---- GIN: mask==all-ones -> neighbor sum == column sum ----
  // conv1: t1 = relu((c + S) @ W1 + b1); h = relu(t1 @ W2 + b2)
  k_colsum<<<dim3(LDIM), 256, 0, stream>>>(bufC, Svec, NCAT, LDIM);
  k_addbcast_bf<<<g1(NCAT * LDIM), 256, 0, stream>>>(bufC, Svec, bfA, NCAT * LDIM, LDIM);
  k_cvt_pad<<<g1(LDIM * GHDIM), 256, 0, stream>>>(ginW1a, bfB, LDIM, GHDIM, LDIM, GHDIM);
  gemm(bfA, LDIM, bfB, GHDIM, bufZ, GHDIM, ginB1a, 1, NCAT, GHDIM, LDIM);
  k_cvt_pad<<<g1(NCAT * GHDIM), 256, 0, stream>>>(bufZ, bfA, NCAT, GHDIM, NCAT, GHDIM);
  k_cvt_pad<<<g1(GHDIM * GHDIM), 256, 0, stream>>>(ginW2a, bfB, GHDIM, GHDIM, GHDIM, GHDIM);
  gemm(bfA, GHDIM, bfB, GHDIM, bufH, GHDIM, ginB2a, 1, NCAT, GHDIM, GHDIM);

  // conv2: out = relu((h + S2) @ W1' + b1') @ W2' + b2'
  k_colsum<<<dim3(GHDIM), 256, 0, stream>>>(bufH, Svec, NCAT, GHDIM);
  k_addbcast_bf<<<g1(NCAT * GHDIM), 256, 0, stream>>>(bufH, Svec, bfA, NCAT * GHDIM, GHDIM);
  k_cvt_pad<<<g1(GHDIM * LDIM), 256, 0, stream>>>(ginW1b, bfB, GHDIM, LDIM, GHDIM, LDIM);
  gemm(bfA, GHDIM, bfB, LDIM, bufZ, LDIM, ginB1b, 1, NCAT, LDIM, GHDIM);
  k_cvt_pad<<<g1(NCAT * LDIM), 256, 0, stream>>>(bufZ, bfA, NCAT, LDIM, NCAT, LDIM);
  k_cvt_pad<<<g1(LDIM * LDIM), 256, 0, stream>>>(ginW2b, bfB, LDIM, LDIM, LDIM, LDIM);
  gemm(bfA, LDIM, bfB, LDIM, out, LDIM, ginB2b, 0, NCAT, LDIM, LDIM);
}
